// RNN_64295660421870
// MI455X (gfx1250) — matrix operand order repaired and checked by reading, not compiled, not run on hardware
//
#include <hip/hip_runtime.h>
#include <cstdint>
#include <cstddef>

// ---------------------------------------------------------------------------
// RNN_RELU (cuDNN-style), 2 layers, S=2048 B=64 H=I=512, fp32 in/out.
// bf16 WMMA (fp32 accumulate) for all matmuls; fp32 activations in HBM.
//   ws layout: [ buf: S*B*512 fp32 (pre/out, reused in place) | wih_bf16 | whh_bf16 ]
// Recurrence: single WGP, h-state in LDS (bf16), pre[t] staged into LDS via
// CDNA5 async-to-LDS loads overlapped with the WMMA K-loop.
// ---------------------------------------------------------------------------

#define SEQ    2048
#define BATCH  64
#define HID    512
#define LAYERS 2

typedef __bf16 bf16;
typedef __attribute__((ext_vector_type(16))) __bf16 v16bf;
typedef __attribute__((ext_vector_type(8)))  __bf16 bf16x8;
typedef __attribute__((ext_vector_type(8)))  float  v8f;
typedef __attribute__((ext_vector_type(4)))  float  f32x4;

// ---- fp32 -> bf16 weight conversion (vectorized x4) -----------------------
__global__ void cvt_bf16_kernel(const float* __restrict__ src,
                                bf16* __restrict__ dst, int n4) {
  int i = blockIdx.x * blockDim.x + threadIdx.x;
  if (i < n4) {
    f32x4 v = ((const f32x4*)src)[i];
#pragma unroll
    for (int k = 0; k < 4; ++k) dst[i * 4 + k] = (bf16)v[k];
  }
}

// ---- pre = A @ W^T + bias  (A fp32 [M,512] row-major, W bf16 [512,512]) ----
// In-place safe (O may == A): block owns 64 rows, reads them fully (K-loop)
// before a workgroup barrier, then writes the same rows.
__global__ __launch_bounds__(1024)
void gemm_wmma_kernel(const float* A,                  // [M, HID] (may alias O)
                      const bf16*  __restrict__ W,     // [HID, HID] bf16
                      const float* __restrict__ bias,  // [HID]
                      float* O) {                      // [M, HID]
  const int lane = threadIdx.x & 31;
  const int wave = threadIdx.x >> 5;   // 0..31
  const int ln15 = lane & 15;
  const int hi   = lane >> 4;          // K-half select (A/B fragment layout)
  const int m    = wave >> 3;          // 0..3  : 16-row strip within 64-row block
  const int ng   = wave & 7;           // 0..7  : 64-col group

  const int r0 = blockIdx.y * 64 + m * 16;
  const int c0 = ng * 64;

  v8f acc[4] = {};

  const float* Arow = A + (size_t)(r0 + ln15) * HID;

  for (int kk = 0; kk < HID; kk += 32) {
    // A fragment: lanes 0-15 rows, elems 0-7 = K[kk+hi*8..], elems 8-15 = K[kk+16+hi*8..]
    v16bf a;
    {
      const f32x4* p0 = (const f32x4*)(Arow + kk + hi * 8);
      f32x4 x0 = p0[0], x1 = p0[1];
      const f32x4* p1 = (const f32x4*)(Arow + kk + 16 + hi * 8);
      f32x4 y0 = p1[0], y1 = p1[1];
#pragma unroll
      for (int i = 0; i < 4; ++i) {
        a[i]      = (bf16)x0[i];
        a[4 + i]  = (bf16)x1[i];
        a[8 + i]  = (bf16)y0[i];
        a[12 + i] = (bf16)y1[i];
      }
    }
    // preload all 4 B fragments, then issue the 4 WMMAs (overlap load latency)
    v16bf bfr[4];
#pragma unroll
    for (int q = 0; q < 4; ++q) {
      const bf16* Wr = W + (size_t)(c0 + q * 16 + ln15) * HID + kk + hi * 8;
      bf16x8 b0 = ((const bf16x8*)Wr)[0];
      bf16x8 b1 = ((const bf16x8*)Wr)[2];
#pragma unroll
      for (int i = 0; i < 8; ++i) { bfr[q][i] = b0[i]; bfr[q][8 + i] = b1[i]; }
    }
#pragma unroll
    for (int q = 0; q < 4; ++q) {
      acc[q] = __builtin_amdgcn_wmma_f32_16x16x32_bf16(
          false, a, false, bfr[q], (short)0, acc[q], false, false);
    }
  }

  __syncthreads();  // all A reads of this 64-row strip done before in-place writes

#pragma unroll
  for (int q = 0; q < 4; ++q) {
    const int c = c0 + q * 16 + ln15;
    const float bv = bias[c];
#pragma unroll
    for (int v = 0; v < 8; ++v) {
      const int r = r0 + hi * 8 + v;  // C/D layout: VGPR v -> M = v + 8*hi
      O[(size_t)r * HID + c] = acc[q][v] + bv;
    }
  }
}

// ---- sequential recurrence: h_t = relu(pre_t + h_{t-1} @ Whh^T + bhh) ------
// Single workgroup (32 waves). h in LDS (bf16, padded rows). Each wave stages
// its private 16x64 fp32 slice of pre[t] into LDS with async-to-LDS loads at
// step start; the copy overlaps the whole WMMA K-loop; one s_wait_asynccnt 0
// (same-wave ordering) before consumption -- no extra barrier needed.
__global__ __launch_bounds__(1024)
void rnn_seq_kernel(float* PRE,                        // [SEQ, BATCH, HID] pre-activations
                    float* OUT,                        // [SEQ, BATCH, HID] outputs (may == PRE)
                    const bf16*  __restrict__ Whh,     // [HID, HID] bf16
                    const float* __restrict__ bhh,     // [HID]
                    const float* __restrict__ hx,      // [BATCH, HID]
                    float* __restrict__ hfin) {        // [BATCH, HID]
  __shared__ bf16  hbuf[BATCH * 520];   // 66.5 KB; +8 bf16 row pad (bank rotation)
  __shared__ float ptile[32 * 1024];    // 128 KB; per-wave 16x64 fp32 slice of pre[t]

  const int lane = threadIdx.x & 31;
  const int wave = threadIdx.x >> 5;
  const int ln15 = lane & 15;
  const int hi   = lane >> 4;
  const int m    = wave >> 3;          // 0..3 : batch-row tile
  const int ng   = wave & 7;           // 0..7 : 64-col group

  // init hidden state from hx (fp32 -> bf16)
  for (int i = threadIdx.x; i < BATCH * HID; i += 1024) {
    hbuf[(i >> 9) * 520 + (i & 511)] = (bf16)hx[i];
  }
  __syncthreads();

  const int arow = m * 16 + ln15;

  float bq[4];
#pragma unroll
  for (int q = 0; q < 4; ++q) bq[q] = bhh[ng * 64 + q * 16 + ln15];

  // async-copy address bases (per lane): 2 rows x 16 B128-chunks per issue
  const unsigned goff0 = (unsigned)(((m * 16 + hi) * HID + ng * 64 + ln15 * 4) *
                                    sizeof(float));
  const unsigned loff0 = (unsigned)(uintptr_t)&ptile[0] +
                         (unsigned)(wave * 4096 + hi * 256 + ln15 * 16);
  const float* wtile = &ptile[wave * 1024];

#pragma unroll 1
  for (int t = 0; t < SEQ; ++t) {
    const float* pre_t = PRE + (size_t)t * (BATCH * HID);
    float*       out_t = OUT + (size_t)t * (BATCH * HID);

    // stage this wave's 16x64 fp32 slice of pre[t] into LDS (ASYNCcnt-tracked);
    // completes under the WMMA K-loop below.
#pragma unroll
    for (int i = 0; i < 8; ++i) {
      unsigned go = goff0 + (unsigned)(i * 2 * HID * sizeof(float));
      unsigned lo = loff0 + (unsigned)(i * 512);
      asm volatile("global_load_async_to_lds_b128 %0, %1, %2"
                   :: "v"(lo), "v"(go), "s"(pre_t) : "memory");
    }

    // prefetch next step's pre-activations toward L2 (speculative)
    __builtin_prefetch(pre_t + BATCH * HID + threadIdx.x * 32, 0, 0);

    v8f acc[4] = {};
    for (int kk = 0; kk < HID; kk += 32) {
      v16bf a;
      {
        bf16x8 ca = *(const bf16x8*)&hbuf[arow * 520 + kk + hi * 8];
        bf16x8 cb = *(const bf16x8*)&hbuf[arow * 520 + kk + 16 + hi * 8];
#pragma unroll
        for (int i = 0; i < 8; ++i) { a[i] = ca[i]; a[8 + i] = cb[i]; }
      }
      v16bf bfr[4];
#pragma unroll
      for (int q = 0; q < 4; ++q) {
        const bf16* Wr = Whh + (size_t)(ng * 64 + q * 16 + ln15) * HID + kk + hi * 8;
        bf16x8 b0 = ((const bf16x8*)Wr)[0];
        bf16x8 b1 = ((const bf16x8*)Wr)[2];
#pragma unroll
        for (int i = 0; i < 8; ++i) { bfr[q][i] = b0[i]; bfr[q][8 + i] = b1[i]; }
      }
#pragma unroll
      for (int q = 0; q < 4; ++q) {
        acc[q] = __builtin_amdgcn_wmma_f32_16x16x32_bf16(
            false, a, false, bfr[q], (short)0, acc[q], false, false);
      }
    }

    // pre[t] slice now needed: wait for this wave's async copies
    asm volatile("s_wait_asynccnt 0x0" ::: "memory");

    // fuse: h = relu(acc + pre + bhh)   (pre read from LDS tile)
    float hv[4][8];
#pragma unroll
    for (int q = 0; q < 4; ++q) {
#pragma unroll
      for (int v = 0; v < 8; ++v) {
        float x = acc[q][v] + wtile[(hi * 8 + v) * 64 + q * 16 + ln15] + bq[q];
        hv[q][v] = x > 0.f ? x : 0.f;
      }
    }

    __syncthreads();  // everyone done reading old h from LDS
#pragma unroll
    for (int q = 0; q < 4; ++q) {
      const int c = ng * 64 + q * 16 + ln15;
#pragma unroll
      for (int v = 0; v < 8; ++v) {
        const int r = m * 16 + hi * 8 + v;
        hbuf[r * 520 + c] = (bf16)hv[q][v];            // next-step operand (bf16)
        out_t[(size_t)r * HID + c] = hv[q][v];         // layer output (fp32)
        if (t == SEQ - 1) hfin[(size_t)r * HID + c] = hv[q][v];
      }
    }
    __syncthreads();  // new h visible before next step
  }
}

// ---------------------------------------------------------------------------
extern "C" void kernel_launch(void* const* d_in, const int* in_sizes, int n_in,
                              void* d_out, int out_size, void* d_ws, size_t ws_size,
                              hipStream_t stream) {
  (void)in_sizes; (void)n_in; (void)out_size; (void)ws_size;

  const float* x    = (const float*)d_in[0];  // [SEQ, BATCH, HID]
  const float* hx   = (const float*)d_in[1];  // [LAYERS, BATCH, HID]
  const float* w_ih = (const float*)d_in[2];  // [LAYERS, HID, HID]
  const float* w_hh = (const float*)d_in[3];  // [LAYERS, HID, HID]
  const float* b_ih = (const float*)d_in[4];  // [LAYERS, HID]
  const float* b_hh = (const float*)d_in[5];  // [LAYERS, HID]
  float* out = (float*)d_out;                 // [SEQ*BATCH*HID] ++ [LAYERS*BATCH*HID]

  const int M = SEQ * BATCH;                  // 131072 rows
  float* buf  = (float*)d_ws;                                          // 256 MB
  bf16* wih_b = (bf16*)((char*)d_ws + (size_t)M * HID * sizeof(float));
  bf16* whh_b = wih_b + (size_t)LAYERS * HID * HID;

  // weights -> bf16 (hot in L2 thereafter)
  {
    const int n4 = (LAYERS * HID * HID) / 4;
    cvt_bf16_kernel<<<(n4 + 255) / 256, 256, 0, stream>>>(w_ih, wih_b, n4);
    cvt_bf16_kernel<<<(n4 + 255) / 256, 256, 0, stream>>>(w_hh, whh_b, n4);
  }

  float* hfin0 = out + (size_t)M * HID;
  float* hfin1 = hfin0 + BATCH * HID;
  const dim3 gemm_grid(1, M / 64);

  // layer 0: pre0 = x @ Wih0^T + b_ih0 ; recurrence in place (buf -> out0)
  gemm_wmma_kernel<<<gemm_grid, 1024, 0, stream>>>(x, wih_b, b_ih, buf);
  rnn_seq_kernel<<<1, 1024, 0, stream>>>(buf, buf, whh_b, b_hh, hx, hfin0);

  // layer 1: pre1 = out0 @ Wih1^T + b_ih1 (in place) ; recurrence -> d_out
  gemm_wmma_kernel<<<gemm_grid, 1024, 0, stream>>>(buf, wih_b + HID * HID,
                                                   b_ih + HID, buf);
  rnn_seq_kernel<<<1, 1024, 0, stream>>>(buf, out, whh_b + HID * HID,
                                         b_hh + HID, hx + BATCH * HID, hfin1);
}
